// DeconvBlockTransformer_57363583206034
// MI455X (gfx1250) — matrix operand order repaired and checked by reading, not compiled, not run on hardware
//
#include <hip/hip_runtime.h>

// ---------------------------------------------------------------------------
// DeconvBlockTransformer for MI455X (gfx1250, wave32, WMMA 16x16x32 f16)
// B=16, D1=256, H1=W1=64, D2=128, H2=W2=128
// ---------------------------------------------------------------------------

typedef _Float16 h16;
typedef __attribute__((ext_vector_type(16))) _Float16 v16h;
typedef __attribute__((ext_vector_type(8)))  _Float16 v8h;
typedef __attribute__((ext_vector_type(8)))  float    v8f;

#define EPSV 1e-5f

// ---------------- WMMA helpers (CDNA5 layouts, wave32) ----------------------

__device__ __forceinline__ v8f wmma_f16(v16h a, v16h b, v8f c) {
  // D = A(16x32) * B(32x16) + C, fp32 accumulate
  return __builtin_amdgcn_wmma_f32_16x16x32_f16(
      false, a, false, b, (short)0, c, false, false);
}

// A fragment: 16x32 f16 tile from row-major storage (k contiguous).
// Layout: lanes 0-15 -> row m=lane, K 0..15; lanes 16-31 -> K 16..31 (matches B)
__device__ __forceinline__ v16h load_a16(const h16* __restrict__ base, int lda) {
  const int lane = threadIdx.x & 31;
  const int m  = lane & 15;
  const int kb = (lane >> 4) << 4;          // 0 or 16
  const h16* p = base + (size_t)m * lda + kb;
  v8h lo = *(const v8h*)(p);
  v8h hi = *(const v8h*)(p + 8);
  return __builtin_shufflevector(lo, hi, 0,1,2,3,4,5,6,7,8,9,10,11,12,13,14,15);
}

// B fragment from B^T stored row-major [N][K] (k contiguous per row).
// Layout used here (16-bit B, same k order as A): lanes 0-15 -> col n=lane, K 0..15; lanes 16-31 -> K 16..31
__device__ __forceinline__ v16h load_bt16(const h16* __restrict__ baseT, int ldb) {
  const int lane = threadIdx.x & 31;
  const int n  = lane & 15;
  const int kb = (lane >> 4) << 4;          // 0 or 16
  return *(const v16h*)(baseT + (size_t)n * ldb + kb);
}

// D (f32 16x16) -> f16 row-major store.  VGPR r: m=r (lanes 0-15) / m=r+8.
__device__ __forceinline__ void store_d_f16(h16* __restrict__ d, int ldd, v8f acc) {
  const int lane = threadIdx.x & 31;
  const int n  = lane & 15;
  const int mb = (lane >> 4) << 3;
#pragma unroll
  for (int r = 0; r < 8; ++r) d[(size_t)(mb + r) * ldd + n] = (h16)acc[r];
}

// ---------------- Positional encodings --------------------------------------

__device__ __forceinline__ float pe_val(int d, int c, int l, int axis) {
  const int dh = d >> 1;
  if (axis == 0 && c < dh)  return (c & 1) ? 1.f : 0.f;          // sin(0)/cos(0)
  if (axis == 1 && c >= dh) return ((c - dh) & 1) ? 1.f : 0.f;
  const int cc = (axis == 0) ? (c - dh) : c;
  const float div = __powf(10000.f, -(float)(cc & ~1) / (float)dh);
  const float a = (float)l * div;
  return (cc & 1) ? __cosf(a) : __sinf(a);
}

__global__ __launch_bounds__(256) void k_pe(float* pe1h, float* pe1w,
                                            float* pe2h, float* pe2w) {
  int idx = blockIdx.x * 256 + threadIdx.x;          // 49152 total
  if (idx >= 2 * 16384 + 2 * 4096) return;
  if (idx < 16384)            pe1h[idx] = pe_val(256, idx >> 6, idx & 63, 0);
  else if (idx < 32768)       { idx -= 16384; pe1w[idx] = pe_val(256, idx >> 6, idx & 63, 1); }
  else if (idx < 36864)       { idx -= 32768; pe2h[idx] = pe_val(128, idx >> 6, idx & 63, 0); }
  else                        { idx -= 36864; pe2w[idx] = pe_val(128, idx >> 6, idx & 63, 1); }
}

// ---------------- Weight preparation ----------------------------------------

// WQK buffer: out[cp][c] = sum_o Wq[o][c] * Wk[o][cp]   (so Bt[n=cp][k=c] fetch works)
__global__ __launch_bounds__(256) void k_wqk(const float* __restrict__ Wq,
                                             const float* __restrict__ Wk,
                                             h16* __restrict__ out) {
  int idx = blockIdx.x * 256 + threadIdx.x;          // 256*256
  if (idx >= 256 * 256) return;
  int c = idx & 255, cp = idx >> 8;
  float s = 0.f;
  for (int o = 0; o < 256; ++o) s += Wq[o * 256 + c] * Wk[o * 256 + cp];
  out[(size_t)cp * 256 + c] = (h16)s;
}

__global__ __launch_bounds__(256) void k_f2h(const float* __restrict__ s,
                                             h16* __restrict__ d, int n) {
  int i = blockIdx.x * 256 + threadIdx.x;
  if (i < n) d[i] = (h16)s[i];
}

// conv weights [O][Ci][3][3] -> tap-major f16 [9][O][Ci]
__global__ __launch_bounds__(256) void k_convw(const float* __restrict__ w,
                                               h16* __restrict__ wT, int O, int Ci) {
  int idx = blockIdx.x * 256 + threadIdx.x;          // O*Ci*9
  if (idx >= O * Ci * 9) return;
  int t = idx % 9, ci = (idx / 9) % Ci, o = idx / (9 * Ci);
  wT[((size_t)t * O + o) * Ci + ci] = (h16)w[idx];
}

// ---------------- Elementwise stages ----------------------------------------

__global__ __launch_bounds__(256) void k_maxpool(const float* __restrict__ b,
                                                 h16* __restrict__ out) {
  size_t idx = (size_t)blockIdx.x * 256 + threadIdx.x;   // 16*128*64*64
  if (idx >= (size_t)16 * 128 * 64 * 64) return;
  int w = idx & 63, h = (idx >> 6) & 63, c = (idx >> 12) & 127, bb = idx >> 19;
  const float* p = b + (((size_t)bb * 128 + c) * 128 + 2 * h) * 128 + 2 * w;
  out[idx] = (h16)fmaxf(fmaxf(p[0], p[1]), fmaxf(p[128], p[129]));
}

// xsT[(b*64+s)][l][c], mode0(h-pass): s=w,l=h ; mode1(w-pass): s=h,l=w
__global__ __launch_bounds__(256) void k_prep_x(const float* __restrict__ x,
                                                const float* __restrict__ pe1h,
                                                const float* __restrict__ pe1w,
                                                h16* __restrict__ xsT, int mode) {
  size_t idx = (size_t)blockIdx.x * 256 + threadIdx.x;   // 1024*64*256
  if (idx >= (size_t)1024 * 64 * 256) return;
  int c = idx & 255, l = (idx >> 8) & 63, g = idx >> 14;
  int b = g >> 6, s = g & 63;
  int hh = mode ? s : l, ww = mode ? l : s;
  float v = x[(((size_t)b * 256 + c) * 64 + hh) * 64 + ww] + pe1h[c * 64 + hh];
  if (mode) v += pe1w[c * 64 + ww];
  xsT[idx] = (h16)v;
}

// bsT[(b*64+s)][l][c]; mode0: src=b_(maxpool)+pe2h[c][l=h]; mode1: src=b2+pe2w[c][l=w]
__global__ __launch_bounds__(256) void k_prep_b(const h16* __restrict__ src,
                                                const float* __restrict__ pe,
                                                h16* __restrict__ bsT, int mode) {
  size_t idx = (size_t)blockIdx.x * 256 + threadIdx.x;   // 1024*64*128
  if (idx >= (size_t)1024 * 64 * 128) return;
  int c = idx & 127, l = (idx >> 7) & 63, g = idx >> 13;
  int b = g >> 6, s = g & 63;
  int hh = mode ? s : l, ww = mode ? l : s;
  float v = (float)src[(((size_t)b * 128 + c) * 64 + hh) * 64 + ww] + pe[c * 64 + l];
  bsT[idx] = (h16)v;
}

// ---------------- Attention GEMMs (per-group, WMMA) -------------------------

// T[g][i][cp] = sum_c xsT[g][i][c] * WQK[c][cp]      M=64 N=256 K=256
__global__ __launch_bounds__(256) void k_attn_T(const h16* __restrict__ xsT,
                                                const h16* __restrict__ wqk,
                                                h16* __restrict__ Tout) {
  const int g = blockIdx.x, wave = threadIdx.x >> 5;
  const h16* A = xsT + (size_t)g * 64 * 256;
  h16* D = Tout + (size_t)g * 64 * 256;
  for (int t = wave; t < 64; t += 8) {
    const int m0 = (t & 3) * 16, n0 = (t >> 2) * 16;
    v8f acc; for (int r = 0; r < 8; ++r) acc[r] = 0.f;
#pragma unroll
    for (int k0 = 0; k0 < 256; k0 += 32)
      acc = wmma_f16(load_a16(A + (size_t)m0 * 256 + k0, 256),
                     load_bt16(wqk + (size_t)n0 * 256 + k0, 256), acc);
    store_d_f16(D + (size_t)m0 * 256 + n0, 256, acc);
  }
}

// S[i][j] = sum_c T[i][c]*xsT[j][c]; fused row softmax; writes attn f32 + f16
__global__ __launch_bounds__(256) void k_attn_S(const h16* __restrict__ T16,
                                                const h16* __restrict__ xsT,
                                                float* __restrict__ attnF,
                                                h16* __restrict__ attnH) {
  __shared__ float S[64][64];
  const int g = blockIdx.x, wave = threadIdx.x >> 5;
  const h16* A  = T16 + (size_t)g * 64 * 256;
  const h16* Bt = xsT + (size_t)g * 64 * 256;
  const int lane = threadIdx.x & 31, n = lane & 15, mb = (lane >> 4) << 3;
  for (int t = wave; t < 16; t += 8) {
    const int m0 = (t & 3) * 16, n0 = (t >> 2) * 16;
    v8f acc; for (int r = 0; r < 8; ++r) acc[r] = 0.f;
#pragma unroll
    for (int k0 = 0; k0 < 256; k0 += 32)
      acc = wmma_f16(load_a16(A + (size_t)m0 * 256 + k0, 256),
                     load_bt16(Bt + (size_t)n0 * 256 + k0, 256), acc);
#pragma unroll
    for (int r = 0; r < 8; ++r) S[m0 + mb + r][n0 + n] = acc[r];
  }
  __syncthreads();
  const int i = threadIdx.x;
  if (i < 64) {
    float mx = -1e30f;
    for (int j = 0; j < 64; ++j) mx = fmaxf(mx, S[i][j]);
    float sum = 0.f;
    for (int j = 0; j < 64; ++j) sum += __expf(S[i][j] - mx);
    const float inv = 1.f / sum;
    float* fr = attnF + (size_t)g * 4096 + i * 64;
    h16*   hr = attnH + (size_t)g * 4096 + i * 64;
    for (int j = 0; j < 64; ++j) {
      float p = __expf(S[i][j] - mx) * inv;
      fr[j] = p; hr[j] = (h16)p;
    }
  }
}

// v[g][o][l] = sum_c Wv[o][c]*bsT[g][l][c]           M=128 N=64 K=128
__global__ __launch_bounds__(256) void k_attn_V(const h16* __restrict__ Wv,
                                                const h16* __restrict__ bsT,
                                                h16* __restrict__ vbuf) {
  const int g = blockIdx.x, wave = threadIdx.x >> 5;
  const h16* Bt = bsT + (size_t)g * 64 * 128;
  h16* D = vbuf + (size_t)g * 128 * 64;
  for (int t = wave; t < 32; t += 8) {
    const int m0 = (t & 7) * 16, n0 = (t >> 3) * 16;
    v8f acc; for (int r = 0; r < 8; ++r) acc[r] = 0.f;
#pragma unroll
    for (int k0 = 0; k0 < 128; k0 += 32)
      acc = wmma_f16(load_a16(Wv + (size_t)m0 * 128 + k0, 128),
                     load_bt16(Bt + (size_t)n0 * 128 + k0, 128), acc);
    store_d_f16(D + (size_t)m0 * 64 + n0, 64, acc);
  }
}

// out[c][i] = sum_j v[c][j]*attn[i][j]               M=128 N=64 K=64
// mode0 (h): ob[c][l][s]  (b2[b][c][h][w], s=w)   mode1 (w): ob[c][s][l]
__global__ __launch_bounds__(256) void k_attn_O(const h16* __restrict__ vbuf,
                                                const h16* __restrict__ attnH,
                                                h16* __restrict__ outb, int mode) {
  const int g = blockIdx.x, wave = threadIdx.x >> 5;
  const int b = g >> 6, s = g & 63;
  const h16* A  = vbuf  + (size_t)g * 128 * 64;
  const h16* Bt = attnH + (size_t)g * 4096;
  h16* ob = outb + (size_t)b * 128 * 64 * 64;
  const int lane = threadIdx.x & 31, n = lane & 15, mb = (lane >> 4) << 3;
  for (int t = wave; t < 32; t += 8) {
    const int m0 = (t & 7) * 16, n0 = (t >> 3) * 16;
    v8f acc; for (int r = 0; r < 8; ++r) acc[r] = 0.f;
    acc = wmma_f16(load_a16(A + (size_t)m0 * 64,      64),
                   load_bt16(Bt + (size_t)n0 * 64,      64), acc);
    acc = wmma_f16(load_a16(A + (size_t)m0 * 64 + 32, 64),
                   load_bt16(Bt + (size_t)n0 * 64 + 32, 64), acc);
#pragma unroll
    for (int r = 0; r < 8; ++r) {
      int c = m0 + mb + r, l = n0 + n;
      size_t a = mode == 0 ? ((size_t)c * 64 + l) * 64 + s
                           : ((size_t)c * 64 + s) * 64 + l;
      ob[a] = (h16)acc[r];
    }
  }
}

// ---------------- Gating / upsampling ---------------------------------------

__global__ __launch_bounds__(256) void k_gating(const float* __restrict__ Wa,
                                                const float* __restrict__ ba,
                                                const float* __restrict__ bng,
                                                const float* __restrict__ bnb,
                                                const h16* __restrict__ outA,
                                                h16* __restrict__ gsig) {
  size_t idx = (size_t)blockIdx.x * 256 + threadIdx.x;   // 16*128*4096
  if (idx >= (size_t)16 * 128 * 4096) return;
  int p = idx & 4095, o = (idx >> 12) & 127, b = idx >> 19;
  const h16* src = outA + (size_t)b * 128 * 4096 + p;
  const float* wr = Wa + (size_t)o * 128;
  float s = 0.f;
  for (int c = 0; c < 128; ++c) s += wr[c] * (float)src[(size_t)c * 4096];
  s = (s + ba[o]) * (bng[o] * rsqrtf(1.f + EPSV)) + bnb[o];
  gsig[idx] = (h16)(1.f / (1.f + __expf(-s)));
}

// b_out NHWC f16: up2x_ac(gsig) * b
__global__ __launch_bounds__(256) void k_bout(const h16* __restrict__ gsig,
                                              const float* __restrict__ bin,
                                              h16* __restrict__ bout) {
  size_t idx = (size_t)blockIdx.x * 256 + threadIdx.x;   // 16*128*128*128
  if (idx >= (size_t)16 * 128 * 128 * 128) return;
  int c = idx & 127, x = (idx >> 7) & 127, y = (idx >> 14) & 127, b = idx >> 21;
  float sy = y * (63.f / 127.f), sx = x * (63.f / 127.f);
  int y0 = (int)sy, x0 = (int)sx;
  int y1 = min(y0 + 1, 63), x1 = min(x0 + 1, 63);
  float fy = sy - y0, fx = sx - x0;
  const h16* gp = gsig + ((size_t)b * 128 + c) * 4096;
  float g00 = gp[y0 * 64 + x0], g01 = gp[y0 * 64 + x1];
  float g10 = gp[y1 * 64 + x0], g11 = gp[y1 * 64 + x1];
  float gv = (g00 * (1 - fx) + g01 * fx) * (1 - fy) + (g10 * (1 - fx) + g11 * fx) * fy;
  float bv = bin[(((size_t)b * 128 + c) * 128 + y) * 128 + x];
  bout[idx] = (h16)(gv * bv);
}

// xup NHWC f16: up2x_ac(x + pe1_h + pe1_w)
__global__ __launch_bounds__(256) void k_xup(const float* __restrict__ x,
                                             const float* __restrict__ pe1h,
                                             const float* __restrict__ pe1w,
                                             h16* __restrict__ xup) {
  size_t idx = (size_t)blockIdx.x * 256 + threadIdx.x;   // 16*128*128*256
  if (idx >= (size_t)16 * 128 * 128 * 256) return;
  int c = idx & 255, xx = (idx >> 8) & 127, y = (idx >> 15) & 127, b = idx >> 22;
  float sy = y * (63.f / 127.f), sx = xx * (63.f / 127.f);
  int y0 = (int)sy, x0 = (int)sx;
  int y1 = min(y0 + 1, 63), x1 = min(x0 + 1, 63);
  float fy = sy - y0, fx = sx - x0;
  const float* xp = x + ((size_t)b * 256 + c) * 4096;
  const float* ph = pe1h + c * 64;
  const float* pw = pe1w + c * 64;
  float v00 = xp[y0 * 64 + x0] + ph[y0] + pw[x0];
  float v01 = xp[y0 * 64 + x1] + ph[y0] + pw[x1];
  float v10 = xp[y1 * 64 + x0] + ph[y1] + pw[x0];
  float v11 = xp[y1 * 64 + x1] + ph[y1] + pw[x1];
  xup[idx] = (h16)((v00 * (1 - fx) + v01 * fx) * (1 - fy) +
                   (v10 * (1 - fx) + v11 * fx) * fy);
}

// ---------------- 3x3 conv: tap-major implicit GEMM -------------------------
// One block = one output row (b, y).  Per 64-channel K-slab, the 3-row
// receptive field (halo-padded) is staged into LDS once via CDNA5 async
// global->LDS copies (ASYNCcnt), then all 8 waves (one M-tile row each)
// consume it: A (weights) loaded once per tap and reused across 8 N-tiles,
// B fragments served from LDS.  Epilogue: bias (+BN) + ReLU.
__global__ __launch_bounds__(256) void k_conv(const h16* __restrict__ in1,
                                              const h16* __restrict__ in2,
                                              int Cin,
                                              const h16* __restrict__ wT,
                                              const float* __restrict__ bias,
                                              const float* __restrict__ bng,
                                              const float* __restrict__ bnb,
                                              h16* __restrict__ outH,
                                              float* __restrict__ outF) {
  __shared__ h16 sIn[3 * 130 * 64];            // 48.75 KB
  const int blk = blockIdx.x;                  // b*128 + y
  const int b = blk >> 7, y = blk & 127;
  const int tid = threadIdx.x;
  const int wave = tid >> 5, lane = tid & 31;
  const int n = lane & 15, mb = (lane >> 4) << 3, kb = (lane >> 4) << 4;
  const int m0 = wave * 16;                    // this wave's out-channel tile
  const int nblk = Cin >> 6;                   // 64-channel K-slabs
  const int Cl = in2 ? (Cin >> 1) : Cin;       // channels per input buffer

  v8f acc[8];
#pragma unroll
  for (int t = 0; t < 8; ++t)
#pragma unroll
    for (int r = 0; r < 8; ++r) acc[t][r] = 0.f;

  for (int kblk = 0; kblk < nblk; ++kblk) {
    const h16* src; int c0;
    if (in2 && kblk >= (nblk >> 1)) { src = in2; c0 = (kblk - (nblk >> 1)) * 64; }
    else                            { src = in1; c0 = kblk * 64; }

    __syncthreads();                           // previous slab fully consumed
    // zero halo columns (x = -1 and x = 128)
    for (int z = tid; z < 384; z += 256) {
      int c = z & 63, rp = z >> 6;             // rp in [0,6)
      int row = rp % 3, px = (rp / 3) ? 129 : 0;
      sIn[(size_t)(row * 130 + px) * 64 + c] = (h16)0.f;
    }
    // async-copy interior: 3 rows x 128 px x 64 ch in 16B chunks (ASYNCcnt)
    for (int q = tid; q < 3072; q += 256) {
      int c8 = q & 7, px = (q >> 3) & 127, row = q >> 10;
      int yy = y + row - 1;
      if ((unsigned)yy < 128u) {
        const h16* g = src + (((size_t)b * 128 + yy) * 128 + px) * Cl + c0 + c8 * 8;
        unsigned loff =
            (unsigned)(uintptr_t)&sIn[((size_t)(row * 130) + px + 1) * 64 + c8 * 8];
        asm volatile("global_load_async_to_lds_b128 %0, %1, off"
                     :: "v"(loff), "v"(g) : "memory");
      }
    }
    asm volatile("s_wait_asynccnt 0x0" ::: "memory");
    __syncthreads();

    for (int dy = -1; dy <= 1; ++dy) {
      const int yy = y + dy;
      if ((unsigned)yy >= 128u) continue;
      const int row = dy + 1;
      for (int dx = -1; dx <= 1; ++dx) {
        const int tap = (dy + 1) * 3 + (dx + 1);
#pragma unroll
        for (int kloc = 0; kloc < 64; kloc += 32) {
          v16h a = load_a16(wT + ((size_t)tap * 128 + m0) * Cin + kblk * 64 + kloc,
                            Cin);
#pragma unroll
          for (int nt = 0; nt < 8; ++nt) {
            const int px = nt * 16 + dx + 1 + n;
            v16h bf = *(const v16h*)(&sIn[((size_t)row * 130 + px) * 64 + kloc + kb]);
            acc[nt] = wmma_f16(a, bf, acc[nt]);
          }
        }
      }
    }
  }

  // epilogue: bias (+BN) + ReLU, NHWC f16 or NCHW f32
  for (int nt = 0; nt < 8; ++nt) {
#pragma unroll
    for (int r = 0; r < 8; ++r) {
      const int o = m0 + mb + r, xo = nt * 16 + n;
      float v = acc[nt][r] + bias[o];
      if (bng) v = v * (bng[o] * rsqrtf(1.f + EPSV)) + bnb[o];
      v = fmaxf(v, 0.f);
      if (outH) outH[(((size_t)b * 128 + y) * 128 + xo) * 128 + o] = (h16)v;
      else      outF[(((size_t)b * 128 + o) * 128 + y) * 128 + xo] = v;
    }
  }
}

// ---------------------------------------------------------------------------

extern "C" void kernel_launch(void* const* d_in, const int* in_sizes, int n_in,
                              void* d_out, int out_size, void* d_ws, size_t ws_size,
                              hipStream_t stream) {
  (void)in_sizes; (void)n_in; (void)out_size; (void)ws_size;
  const float* x     = (const float*)d_in[0];
  const float* bIn   = (const float*)d_in[1];
  const float* Wq_h  = (const float*)d_in[2];
  const float* Wk_h  = (const float*)d_in[3];
  const float* Wv_h  = (const float*)d_in[4];
  const float* Wq_w  = (const float*)d_in[5];
  const float* Wk_w  = (const float*)d_in[6];
  const float* Wv_w  = (const float*)d_in[7];
  const float* Wa    = (const float*)d_in[8];
  const float* ba    = (const float*)d_in[9];
  const float* bn_ag = (const float*)d_in[10];
  const float* bn_ab = (const float*)d_in[11];
  const float* Wup   = (const float*)d_in[12];
  const float* bup   = (const float*)d_in[13];
  const float* Wc1   = (const float*)d_in[14];
  const float* bc1   = (const float*)d_in[15];
  const float* bn1g  = (const float*)d_in[16];
  const float* bn1b  = (const float*)d_in[17];
  const float* Wc2   = (const float*)d_in[18];
  const float* bc2   = (const float*)d_in[19];
  const float* bn2g  = (const float*)d_in[20];
  const float* bn2b  = (const float*)d_in[21];

  float* yOut   = (float*)d_out;                       // [16,128,128,128]
  float* attnHF = yOut + (size_t)16 * 128 * 128 * 128; // [1024,64,64]
  float* attnWF = attnHF + (size_t)1024 * 64 * 64;

  char* ws = (char*)d_ws;
  size_t off = 0;
  auto take = [&](size_t bytes) -> void* {
    void* p = ws + off; off += (bytes + 255) & ~(size_t)255; return p;
  };
  float* pe1h = (float*)take(256 * 64 * 4);
  float* pe1w = (float*)take(256 * 64 * 4);
  float* pe2h = (float*)take(128 * 64 * 4);
  float* pe2w = (float*)take(128 * 64 * 4);
  h16* WQKh  = (h16*)take((size_t)256 * 256 * 2);
  h16* WQKw  = (h16*)take((size_t)256 * 256 * 2);
  h16* Wvh16 = (h16*)take((size_t)128 * 128 * 2);
  h16* Wvw16 = (h16*)take((size_t)128 * 128 * 2);
  h16* wupT  = (h16*)take((size_t)9 * 128 * 256 * 2);
  h16* wc1T  = (h16*)take((size_t)9 * 128 * 256 * 2);
  h16* wc2T  = (h16*)take((size_t)9 * 128 * 128 * 2);
  h16* outA16 = (h16*)take((size_t)16 * 128 * 4096 * 2);
  h16* gsig16 = (h16*)take((size_t)16 * 128 * 4096 * 2);
  h16* xu16   = (h16*)take((size_t)16 * 128 * 128 * 128 * 2);
  h16* bout16 = (h16*)take((size_t)16 * 128 * 128 * 128 * 2);
  // Attention scratch region (later overlaid by xup, then by y1):
  char* region = ws + off;
  h16* b_16  = (h16*)take((size_t)16 * 128 * 4096 * 2);
  h16* xsT   = (h16*)take((size_t)1024 * 64 * 256 * 2);
  h16* bsT   = (h16*)take((size_t)1024 * 64 * 128 * 2);
  h16* T16   = (h16*)take((size_t)1024 * 64 * 256 * 2);
  h16* attn16 = (h16*)take((size_t)1024 * 64 * 64 * 2);
  h16* v16b  = (h16*)take((size_t)1024 * 128 * 64 * 2);
  h16* b2_16 = (h16*)take((size_t)16 * 128 * 4096 * 2);
  h16* xup16 = (h16*)region;    // 134MB overlay (region = 142.6MB, attn dead)
  h16* y1_16 = (h16*)region;    // 67MB overlay (xup dead after up-conv)

  const int TB = 256;
  auto blk = [](size_t n) { return (unsigned)((n + 255) / 256); };

  // --- constants / weights ---
  k_pe<<<blk(49152), TB, 0, stream>>>(pe1h, pe1w, pe2h, pe2w);
  k_wqk<<<blk(65536), TB, 0, stream>>>(Wq_h, Wk_h, WQKh);
  k_wqk<<<blk(65536), TB, 0, stream>>>(Wq_w, Wk_w, WQKw);
  k_f2h<<<blk(16384), TB, 0, stream>>>(Wv_h, Wvh16, 16384);
  k_f2h<<<blk(16384), TB, 0, stream>>>(Wv_w, Wvw16, 16384);
  k_convw<<<blk((size_t)128 * 256 * 9), TB, 0, stream>>>(Wup, wupT, 128, 256);
  k_convw<<<blk((size_t)128 * 256 * 9), TB, 0, stream>>>(Wc1, wc1T, 128, 256);
  k_convw<<<blk((size_t)128 * 128 * 9), TB, 0, stream>>>(Wc2, wc2T, 128, 128);
  k_maxpool<<<blk((size_t)16 * 128 * 4096), TB, 0, stream>>>(bIn, b_16);

  // --- height-axis attention ---
  k_prep_x<<<blk((size_t)1024 * 64 * 256), TB, 0, stream>>>(x, pe1h, pe1w, xsT, 0);
  k_prep_b<<<blk((size_t)1024 * 64 * 128), TB, 0, stream>>>(b_16, pe2h, bsT, 0);
  k_attn_T<<<1024, TB, 0, stream>>>(xsT, WQKh, T16);
  k_attn_S<<<1024, TB, 0, stream>>>(T16, xsT, attnHF, attn16);
  k_attn_V<<<1024, TB, 0, stream>>>(Wvh16, bsT, v16b);
  k_attn_O<<<1024, TB, 0, stream>>>(v16b, attn16, b2_16, 0);

  // --- width-axis attention ---
  k_prep_x<<<blk((size_t)1024 * 64 * 256), TB, 0, stream>>>(x, pe1h, pe1w, xsT, 1);
  k_prep_b<<<blk((size_t)1024 * 64 * 128), TB, 0, stream>>>(b2_16, pe2w, bsT, 1);
  k_attn_T<<<1024, TB, 0, stream>>>(xsT, WQKw, T16);
  k_attn_S<<<1024, TB, 0, stream>>>(T16, xsT, attnWF, attn16);
  k_attn_V<<<1024, TB, 0, stream>>>(Wvw16, bsT, v16b);
  k_attn_O<<<1024, TB, 0, stream>>>(v16b, attn16, outA16, 1);

  // --- gating + skip path ---
  k_gating<<<blk((size_t)16 * 128 * 4096), TB, 0, stream>>>(Wa, ba, bn_ag, bn_ab,
                                                            outA16, gsig16);
  k_bout<<<blk((size_t)16 * 128 * 16384), TB, 0, stream>>>(gsig16, bIn, bout16);

  // --- up-conv path ---
  k_xup<<<blk((size_t)16 * 256 * 16384), TB, 0, stream>>>(x, pe1h, pe1w, xup16);
  k_conv<<<2048, TB, 0, stream>>>(xup16, nullptr, 256, wupT, bup,
                                  nullptr, nullptr, xu16, nullptr);

  // --- conv1 (concat [xu, b_out]) -> BN -> ReLU ---
  k_conv<<<2048, TB, 0, stream>>>(xu16, bout16, 256, wc1T, bc1,
                                  bn1g, bn1b, y1_16, nullptr);

  // --- conv2 -> BN -> ReLU -> y (fp32 NCHW) ---
  k_conv<<<2048, TB, 0, stream>>>(y1_16, nullptr, 128, wc2T, bc2,
                                  bn2g, bn2b, nullptr, yOut);
}
